// CLIPVisionTower_VisionZip_EXP_44178033607150
// MI455X (gfx1250) — matrix operand order repaired and checked by reading, not compiled, not run on hardware
//
#include <hip/hip_runtime.h>
#include <hip/hip_fp16.h>

typedef __attribute__((ext_vector_type(16))) _Float16 v16h;
typedef __attribute__((ext_vector_type(8)))  _Float16 v8h;
typedef __attribute__((ext_vector_type(8)))  float    v8f;

#define EPSF   1e-12f
#define BB     16
#define HH     16
#define LL     577
#define LW     576     // L - 1 (scored tokens)
#define CK     64
#define CH     1024
#define DOMK   54
#define D1     55      // dominant_num + 1
#define KCTX   10
#define NFILT  522     // L - D1
#define STEP   52      // NFILT / KCTX
#define NREM   512     // NFILT - KCTX
#define OUTTOK 65      // D1 + KCTX

// ---------------------------------------------------------------- kernel 1
// s_attn[b][j] = mean over heads of attn[b,h,0,1+j]  (reads 0.6MB of 341MB)
__global__ void k_sattn(const float* __restrict__ attn, float* __restrict__ sattn) {
    int g = blockIdx.x * blockDim.x + threadIdx.x;
    if (g >= BB * LW) return;
    int b = g / LW, j = g % LW;
    const float* p = attn + (size_t)(b * HH) * (size_t)LL * (size_t)LL + 1 + j;
    float s = 0.f;
#pragma unroll
    for (int h = 0; h < HH; ++h) s += p[(size_t)h * LL * LL];
    sattn[g] = s * (1.0f / HH);
}

// ---------------------------------------------------------------- kernel 2
// per token: H_feat (softmax tau=0.2 entropy over 64) + z = x/||x|| in f16
__global__ void k_feat(const float* __restrict__ metric,
                       _Float16* __restrict__ zh, float* __restrict__ hfeat) {
    int wave = (blockIdx.x * blockDim.x + threadIdx.x) >> 5;
    int lane = threadIdx.x & 31;
    if (wave >= BB * LW) return;
    int b = wave / LW, t = wave % LW;
    const float* row = metric + ((size_t)b * LL + (t + 1)) * CK;
    float x0 = row[lane], x1 = row[lane + 32];
    const float inv_tau = 5.0f;  // 1/0.2
    float a0 = x0 * inv_tau, a1 = x1 * inv_tau;
    float ss = x0 * x0 + x1 * x1;
    float mx = fmaxf(a0, a1);
#pragma unroll
    for (int m = 1; m < 32; m <<= 1) {
        ss += __shfl_xor(ss, m, 32);
        mx  = fmaxf(mx, __shfl_xor(mx, m, 32));
    }
    float e0 = __expf(a0 - mx), e1 = __expf(a1 - mx);
    float S = e0 + e1;
    float T = e0 * (a0 - mx) + e1 * (a1 - mx);
#pragma unroll
    for (int m = 1; m < 32; m <<= 1) {
        S += __shfl_xor(S, m, 32);
        T += __shfl_xor(T, m, 32);
    }
    if (lane == 0)
        hfeat[wave] = (logf(S) - T / S) * (1.0f / logf(64.0f));
    float inv_n = 1.0f / (sqrtf(ss) + EPSF);
    _Float16* zr = zh + (size_t)wave * CK;
    zr[lane]      = (_Float16)(x0 * inv_n);
    zr[lane + 32] = (_Float16)(x1 * inv_n);
}

// ---------------------------------------------------------------- kernel 3
// sim = z z^T via V_WMMA_F32_16X16X32_F16, fused with diag-masked softmax
// (tau=0.1) entropy -> I_mut. sim never touches HBM.
// Since ||z_row||=1, sim<=1 => a=sim/tau<=10: entropy is shift-invariant, so
// use a FIXED shift (10.0) instead of a running max -> branch-free inner
// loop (1 exp + 2 fma per element) and plain-sum lane reduction.
__device__ inline v16h frag_a(const _Float16* zb, int row, int half, int kh) {
    // A 16x32 layout: halves 0..7 -> K = kh*32 + half*8 + i ; 8..15 -> +16
    const _Float16* p = zb + (size_t)row * CK + kh * 32 + half * 8;
    v8h lo = *(const v8h*)(p);
    v8h hi = *(const v8h*)(p + 16);
    v16h f;
#pragma unroll
    for (int i = 0; i < 8; ++i) { f[i] = lo[i]; f[i + 8] = hi[i]; }
    return f;
}
__device__ inline v16h frag_b(const _Float16* zb, int row, int half, int kh) {
    // B 32x16 layout: halves 0..7 -> K = kh*32 + half*8 + i ; 8..15 -> +16
    const _Float16* p = zb + (size_t)row * CK + kh * 32 + half * 8;
    v8h lo = *(const v8h*)(p);
    v8h hi = *(const v8h*)(p + 16);
    v16h f;
#pragma unroll
    for (int i = 0; i < 8; ++i) { f[i] = lo[i]; f[i + 8] = hi[i]; }
    return f;
}

__global__ void __launch_bounds__(32) k_sim(const _Float16* __restrict__ zh,
                                            float* __restrict__ imut) {
    const int b = blockIdx.y;
    const int rt = blockIdx.x;                 // row tile 0..35
    const int lane = threadIdx.x & 31;
    const int n = lane & 15, half = lane >> 4;
    const _Float16* zb = zh + (size_t)b * LW * CK;
    const int row_base = rt * 16;

    const v16h a0 = frag_a(zb, row_base + n, half, 0);
    const v16h a1 = frag_a(zb, row_base + n, half, 1);

    float S[8], T[8];
#pragma unroll
    for (int v = 0; v < 8; ++v) { S[v] = 0.f; T[v] = 0.f; }

    const float inv_tau = 10.0f;   // 1/0.1
    const float SHIFT   = 10.0f;   // fixed shift: a <= 10 always

    // software pipeline: B fragments for tile ct+1 are loaded while the
    // exp/fma epilogue of tile ct runs, hiding the s_wait before the WMMAs.
    v16h b0 = frag_b(zb, n, half, 0);
    v16h b1 = frag_b(zb, n, half, 1);
    for (int ct = 0; ct < LW / 16; ++ct) {
        v8f c = {};
        c = __builtin_amdgcn_wmma_f32_16x16x32_f16(false, a0, false, b0,
                                                   (short)0, c, false, false);
        c = __builtin_amdgcn_wmma_f32_16x16x32_f16(false, a1, false, b1,
                                                   (short)0, c, false, false);
        if (ct + 1 < LW / 16) {
            b0 = frag_b(zb, (ct + 1) * 16 + n, half, 0);
            b1 = frag_b(zb, (ct + 1) * 16 + n, half, 1);
        }
        const int col = ct * 16 + n;
#pragma unroll
        for (int v = 0; v < 8; ++v) {
            const int r = row_base + v + 8 * half;
            // diag mask: exp((-1e10-10)) == 0, drops out exactly
            float am = (col == r) ? (-1.0e10f - SHIFT)
                                  : fmaf(c[v], inv_tau, -SHIFT);
            float e = __expf(am);
            S[v] += e;
            T[v] = fmaf(e, am, T[v]);
        }
    }
    const float inv_logN = 1.0f / logf((float)LW);
#pragma unroll
    for (int v = 0; v < 8; ++v) {
        float S1 = S[v], T1 = T[v];
#pragma unroll
        for (int m = 1; m < 16; m <<= 1) {   // plain-sum butterfly, half-wave
            S1 += __shfl_xor(S1, m, 32);
            T1 += __shfl_xor(T1, m, 32);
        }
        if (n == 0) {
            float H = (logf(S1) - T1 / S1) * inv_logN;
            imut[(size_t)b * LW + row_base + v + 8 * half] = 1.0f - H;
        }
    }
}

// ---------------------------------------------------------------- kernel 4
__device__ float block_sum576(float v, float* sbuf) {
    int tid = threadIdx.x, lane = tid & 31, w = tid >> 5;  // 18 waves
#pragma unroll
    for (int m = 1; m < 32; m <<= 1) v += __shfl_xor(v, m, 32);
    if (lane == 0) sbuf[w] = v;
    __syncthreads();
    if (tid == 0) {
        float s = 0.f;
        for (int i = 0; i < 18; ++i) s += sbuf[i];
        sbuf[0] = s;
    }
    __syncthreads();
    float out = sbuf[0];
    __syncthreads();
    return out;
}

__global__ void __launch_bounds__(576) k_scores(const float* __restrict__ sattn,
                                                const float* __restrict__ hfeat,
                                                const float* __restrict__ imut,
                                                float* __restrict__ scores) {
    __shared__ float sbuf[32];
    int b = blockIdx.x, j = threadIdx.x;
    float vals[3] = { sattn[b * LW + j], hfeat[b * LW + j], imut[b * LW + j] };
    const float wgt[3] = { 1.0f, 0.4f, 0.6f };
    float out = 0.f;
#pragma unroll
    for (int k = 0; k < 3; ++k) {
        float mu  = block_sum576(vals[k], sbuf) * (1.0f / LW);
        float d   = vals[k] - mu;
        float var = block_sum576(d * d, sbuf) * (1.0f / (LW - 1));  // ddof=1
        out += wgt[k] * d / (sqrtf(var) + EPSF);
    }
    scores[b * LW + j] = out;
}

// ---------------------------------------------------------------- kernel 5
// top-54 (jax tie-break: lower index on equal value), then build the sorted
// dominant / target / remain original-token index sets.
__global__ void __launch_bounds__(576) k_topk(const float* __restrict__ scores,
                                              float* __restrict__ out_tail,
                                              int* __restrict__ dom,
                                              int* __restrict__ tgt_org,
                                              int* __restrict__ remain_org) {
    __shared__ float ov[LW];
    __shared__ float wv[1024];
    __shared__ int   wi[1024];
    __shared__ unsigned char taken[LW];
    __shared__ unsigned char sel[LL];
    __shared__ int top[DOMK];
    int b = blockIdx.x, tid = threadIdx.x;
    ov[tid] = scores[b * LW + tid];
    taken[tid] = 0;
    sel[tid] = 0;
    if (tid == 0) { sel[576] = 0; sel[0] = 1; }
    __syncthreads();
    for (int k = 0; k < DOMK; ++k) {
        for (int j = tid; j < 1024; j += 576) {
            wv[j] = (j < LW && !taken[j]) ? ov[j] : -3.4e38f;
            wi[j] = j;
        }
        __syncthreads();
        for (int s = 512; s > 0; s >>= 1) {
            for (int j = tid; j < s; j += 576) {
                float v1 = wv[j], v2 = wv[j + s];
                int   i1 = wi[j], i2 = wi[j + s];
                if (v2 > v1 || (v2 == v1 && i2 < i1)) { wv[j] = v2; wi[j] = i2; }
            }
            __syncthreads();
        }
        if (tid == 0) {
            int w = wi[0];
            top[k] = w + 1;       // token index
            taken[w] = 1;
            sel[w + 1] = 1;
        }
        __syncthreads();
    }
    if (tid == 0) {
        float* dst = out_tail + b * D1;   // all_indices (top-k order, 0 first)
        dst[0] = 0.0f;
        for (int k = 0; k < DOMK; ++k) dst[1 + k] = (float)top[k];
        int dc = 0, fc = 0, rc = 0;       // ascending = stable argsort of mask
        for (int t = 0; t < LL; ++t) {
            if (sel[t]) dom[b * D1 + (dc++)] = t;
            else {
                int i = fc++;
                if ((i % STEP) == 0 && (i / STEP) < KCTX)
                    tgt_org[b * KCTX + i / STEP] = t;
                else
                    remain_org[b * NREM + (rc++)] = t;
            }
        }
    }
}

// ---------------------------------------------------------------- kernel 6
__global__ void __launch_bounds__(256) k_dom(const float* __restrict__ hidden,
                                             const int* __restrict__ dom,
                                             float* __restrict__ out) {
    int b = blockIdx.y, d = blockIdx.x, tid = threadIdx.x;
    int tok = dom[b * D1 + d];
    const float4* src = (const float4*)(hidden + ((size_t)b * LL + tok) * CH);
    float4* dst = (float4*)(out + ((size_t)b * OUTTOK + d) * CH);
    dst[tid] = src[tid];
}

// ---------------------------------------------------------------- kernel 7
__global__ void __launch_bounds__(256) k_assign(const float* __restrict__ metric,
                                                const int* __restrict__ tgt_org,
                                                const int* __restrict__ remain_org,
                                                int* __restrict__ assign,
                                                int* __restrict__ counts) {
    __shared__ float tn[KCTX][CK];
    __shared__ float tscale[KCTX];
    __shared__ int cnt[KCTX];
    int b = blockIdx.x, tid = threadIdx.x;
    if (tid < KCTX) cnt[tid] = 0;
    for (int i = tid; i < KCTX * CK; i += 256) {
        int t = i / CK, c = i % CK;
        tn[t][c] = metric[((size_t)b * LL + tgt_org[b * KCTX + t]) * CK + c];
    }
    __syncthreads();
    if (tid < KCTX) {
        float ss = 0.f;
        for (int c = 0; c < CK; ++c) { float v = tn[tid][c]; ss += v * v; }
        tscale[tid] = 1.0f / (sqrtf(ss) + EPSF);
    }
    __syncthreads();
    for (int i = tid; i < KCTX * CK; i += 256) tn[i / CK][i % CK] *= tscale[i / CK];
    __syncthreads();
    for (int m = tid; m < NREM; m += 256) {
        const float* row = metric + ((size_t)b * LL + remain_org[b * NREM + m]) * CK;
        float d[KCTX];
#pragma unroll
        for (int k = 0; k < KCTX; ++k) d[k] = 0.f;
        for (int c = 0; c < CK; ++c) {
            float x = row[c];
#pragma unroll
            for (int k = 0; k < KCTX; ++k) d[k] += x * tn[k][c];
        }
        int best = 0; float bv = d[0];   // argmax invariant to row norm
#pragma unroll
        for (int k = 1; k < KCTX; ++k)
            if (d[k] > bv) { bv = d[k]; best = k; }
        assign[b * NREM + m] = best;
        atomicAdd(&cnt[best], 1);
    }
    __syncthreads();
    if (tid < KCTX) counts[b * KCTX + tid] = cnt[tid];
}

// ---------------------------------------------------------------- kernel 8
__global__ void __launch_bounds__(256) k_ctx(const float* __restrict__ hidden,
                                             const int* __restrict__ tgt_org,
                                             const int* __restrict__ remain_org,
                                             const int* __restrict__ assign,
                                             const int* __restrict__ counts,
                                             float* __restrict__ out) {
    __shared__ int amem[NREM];
    __shared__ int rmem[NREM];
    int b = blockIdx.y, t = blockIdx.x, tid = threadIdx.x;
    for (int i = tid; i < NREM; i += 256) {
        amem[i] = assign[b * NREM + i];
        rmem[i] = remain_org[b * NREM + i];
    }
    __syncthreads();
    float4 acc = ((const float4*)(hidden +
        ((size_t)b * LL + tgt_org[b * KCTX + t]) * CH))[tid];
    float4 s = make_float4(0.f, 0.f, 0.f, 0.f);
    for (int m = 0; m < NREM; ++m) {       // uniform branch -> scalar skip
        if (amem[m] == t) {
            float4 hv = ((const float4*)(hidden +
                ((size_t)b * LL + rmem[m]) * CH))[tid];
            s.x += hv.x; s.y += hv.y; s.z += hv.z; s.w += hv.w;
        }
    }
    float inv = 1.0f / fmaxf((float)counts[b * KCTX + t], 1.0f);
    acc.x += s.x * inv; acc.y += s.y * inv; acc.z += s.z * inv; acc.w += s.w * inv;
    ((float4*)(out + ((size_t)b * OUTTOK + D1 + t) * CH))[tid] = acc;
}

// ---------------------------------------------------------------- launcher
extern "C" void kernel_launch(void* const* d_in, const int* in_sizes, int n_in,
                              void* d_out, int out_size, void* d_ws, size_t ws_size,
                              hipStream_t stream) {
    const float* attn   = (const float*)d_in[0];
    const float* hidden = (const float*)d_in[1];
    const float* metric = (const float*)d_in[2];
    // dominant_num=54 / contextual_num=10 fixed by the reference shapes.

    char* w = (char*)d_ws;
    _Float16* zh = (_Float16*)w;  w += (size_t)BB * LW * CK * sizeof(_Float16);
    float* sattn  = (float*)w;    w += (size_t)BB * LW * sizeof(float);
    float* hfeat  = (float*)w;    w += (size_t)BB * LW * sizeof(float);
    float* imut   = (float*)w;    w += (size_t)BB * LW * sizeof(float);
    float* scores = (float*)w;    w += (size_t)BB * LW * sizeof(float);
    int* dom      = (int*)w;      w += (size_t)BB * D1 * sizeof(int);
    int* tgt_org  = (int*)w;      w += (size_t)BB * KCTX * sizeof(int);
    int* remain_o = (int*)w;      w += (size_t)BB * NREM * sizeof(int);
    int* assign   = (int*)w;      w += (size_t)BB * NREM * sizeof(int);
    int* counts   = (int*)w;      w += (size_t)BB * KCTX * sizeof(int);

    float* out = (float*)d_out;
    float* out_tail = out + (size_t)BB * OUTTOK * CH;  // all_indices region

    k_sattn<<<dim3((BB * LW + 255) / 256), dim3(256), 0, stream>>>(attn, sattn);
    k_feat<<<dim3(BB * LW / 8), dim3(256), 0, stream>>>(metric, zh, hfeat);
    k_sim<<<dim3(LW / 16, BB), dim3(32), 0, stream>>>(zh, imut);
    k_scores<<<dim3(BB), dim3(576), 0, stream>>>(sattn, hfeat, imut, scores);
    k_topk<<<dim3(BB), dim3(576), 0, stream>>>(scores, out_tail, dom, tgt_org, remain_o);
    k_dom<<<dim3(D1, BB), dim3(256), 0, stream>>>(hidden, dom, out);
    k_assign<<<dim3(BB), dim3(256), 0, stream>>>(metric, tgt_org, remain_o, assign, counts);
    k_ctx<<<dim3(KCTX, BB), dim3(256), 0, stream>>>(hidden, tgt_org, remain_o, assign, counts, out);
}
